// YOLOLoss_755914244221
// MI455X (gfx1250) — compile-verified, matrix operand order checked by reading, not run on hardware
//
#include <hip/hip_runtime.h>
#include <hip/hip_bf16.h>

typedef __attribute__((ext_vector_type(16))) _Float16 v16h;
typedef __attribute__((ext_vector_type(8)))  float    v8f;

#define NUM_CLASSES 80
#define GRID_W 76
#define GRID_WH (76*76)
#define A_PER 3
#define NT 20
#define EPSF 1e-7f
#define IGN_THR 0.5f
#define INV_PI2_4 0.405284734569f   // 4/pi^2

__constant__ float c_anchors[9][2] = {
  {12.f,16.f},{19.f,36.f},{40.f,28.f},{36.f,75.f},{76.f,55.f},
  {72.f,146.f},{142.f,110.f},{192.f,243.f},{459.f,401.f}
};

__device__ __forceinline__ float sigmoidf(float x){ return 1.0f/(1.0f+__expf(-x)); }
// BCE with clipped prob, target==1 and target==0 specializations
__device__ __forceinline__ float bce1(float p){ p = fminf(fmaxf(p,EPSF),1.0f-EPSF); return -__logf(p); }
__device__ __forceinline__ float bce0(float p){ p = fminf(fmaxf(p,EPSF),1.0f-EPSF); return -__logf(1.0f-p); }

// ---------------------------------------------------------------------------
// Main kernel: one thread per grid cell pixel of one (batch, anchor) plane.
// grid.x tiles the 76x76 plane, grid.y = B*A planes.
// Reads only channels 0..4 per cell (5.5 MB total); class channels only for
// the ~320 positive cells. Block loss reduced per-wave with V_WMMA (f16 A/B,
// f32 accumulate) against an all-ones B matrix (row sums = lane partials).
// ---------------------------------------------------------------------------
__global__ void __launch_bounds__(256)
yolo_main(const float* __restrict__ in, const float* __restrict__ tgt,
          float* __restrict__ out)
{
  __shared__ float sgx[NT], sgy[NT], sgw[NT], sgh[NT], sbls[NT];
  __shared__ int   sai[NT], sgi[NT], sgj[NT], scls[NT];
  __shared__ float waveSum[8];

  const int plane = blockIdx.y;          // 0..47
  const int b = plane / A_PER;
  const int a = plane % A_PER;
  const int tid = threadIdx.x;

  // --- per-batch target preprocessing (threads 0..19) ---
  if (tid < NT) {
    const float* t = tgt + (size_t)(b*NT + tid)*5;
    float gx = t[0]*(float)GRID_W, gy = t[1]*(float)GRID_W;
    float gw = t[2]*(float)GRID_W, gh = t[3]*(float)GRID_W;
    int gi = (int)floorf(gx), gj = (int)floorf(gy);
    int best = 0; float bestIou = -1.0f;
    #pragma unroll
    for (int k=0;k<9;k++){
      float aw = c_anchors[k][0]*0.125f, ah = c_anchors[k][1]*0.125f;  // /stride(8)
      float inter = fminf(gw,aw)*fminf(gh,ah);
      float uni   = gw*gh + aw*ah - inter;
      float iou   = inter / fmaxf(uni, 1e-6f);
      if (iou > bestIou){ bestIou = iou; best = k; }
    }
    bool valid = (best >= 6) && (best < 9) && (gj < GRID_W) && (gi < GRID_W);
    sai[tid] = valid ? (best - 6) : A_PER;    // A_PER = never matches
    sgx[tid]=gx; sgy[tid]=gy; sgw[tid]=gw; sgh[tid]=gh;
    sgi[tid]=gi; sgj[tid]=gj;
    sbls[tid] = 2.0f - t[2]*t[3];             // box_loss_scale
    scls[tid] = (int)t[4];
  }
  if (tid < 8) waveSum[tid] = 0.0f;
  __syncthreads();

  const int idx = blockIdx.x*blockDim.x + tid;   // pixel in plane
  float partial = 0.0f;

  if (idx < GRID_WH) {
    const int i = idx % GRID_W;
    const int j = idx / GRID_W;
    const float* base = in + (size_t)(b*255 + a*85)*GRID_WH + idx;

    float rx = base[0];
    float ry = base[(size_t)1*GRID_WH];
    float rw = base[(size_t)2*GRID_WH];
    float rh = base[(size_t)3*GRID_WH];
    float rc = base[(size_t)4*GRID_WH];

    float px = sigmoidf(rx), py = sigmoidf(ry);
    float conf = sigmoidf(rc);
    float aw = c_anchors[6+a][0]*0.125f, ah = c_anchors[6+a][1]*0.125f;
    float bx = px + (float)i, by = py + (float)j;
    float bw = __expf(rw)*aw, bh = __expf(rh)*ah;

    float bminx = bx - bw*0.5f, bmaxx = bx + bw*0.5f;
    float bminy = by - bh*0.5f, bmaxy = by + bh*0.5f;
    float areaB = bw*bh;

    // iou_max over all 20 GT boxes -> ignore mask
    float ioumax = 0.0f;
    #pragma unroll 4
    for (int n=0;n<NT;n++){
      float ghw = sgw[n]*0.5f, ghh = sgh[n]*0.5f;
      float iw = fmaxf(fminf(bmaxx, sgx[n]+ghw) - fmaxf(bminx, sgx[n]-ghw), 0.0f);
      float ih = fmaxf(fminf(bmaxy, sgy[n]+ghh) - fmaxf(bminy, sgy[n]-ghh), 0.0f);
      float inter = iw*ih;
      float uni = areaB + sgw[n]*sgh[n] - inter;
      ioumax = fmaxf(ioumax, inter / fmaxf(uni, 1e-6f));
    }

    // positive cell? (last matching target wins, mimicking scatter-set)
    int hit = -1;
    #pragma unroll 4
    for (int n=0;n<NT;n++)
      if (sai[n]==a && sgj[n]==j && sgi[n]==i) hit = n;

    if (hit >= 0) {
      // conf BCE, target 1 (no_obj already 0 here)
      partial += bce1(conf);

      // class BCE over 80 classes (positive cells only)
      const float* cbase = base + (size_t)5*GRID_WH;
      __builtin_prefetch(cbase, 0, 1);   // global_prefetch_b8 for class plane
      int tc = scls[hit];
      for (int k=0;k<NUM_CLASSES;k++){
        float pc = sigmoidf(cbase[(size_t)k*GRID_WH]);
        partial += (k==tc) ? bce1(pc) : bce0(pc);
      }

      // CIoU localization loss
      float gx=sgx[hit], gy=sgy[hit], gw=sgw[hit], gh=sgh[hit];
      float gminx = gx-gw*0.5f, gmaxx = gx+gw*0.5f;
      float gminy = gy-gh*0.5f, gmaxy = gy+gh*0.5f;
      float iw = fmaxf(fminf(bmaxx,gmaxx)-fmaxf(bminx,gminx), 0.0f);
      float ih = fmaxf(fminf(bmaxy,gmaxy)-fmaxf(bminy,gminy), 0.0f);
      float inter = iw*ih;
      float uni = areaB + gw*gh - inter;
      float iou = inter / fmaxf(uni, 1e-6f);
      float dx = bx-gx, dy = by-gy;
      float cd2 = dx*dx + dy*dy;
      float ew = fmaxf(fmaxf(bmaxx,gmaxx)-fminf(bminx,gminx), 0.0f);
      float eh = fmaxf(fmaxf(bmaxy,gmaxy)-fminf(bminy,gminy), 0.0f);
      float ed = ew*ew + eh*eh;
      float ciou = iou - cd2 / fmaxf(ed, 1e-6f);
      float dv = atanf(bw/fmaxf(bh,1e-6f)) - atanf(gw/fmaxf(gh,1e-6f));
      float v  = INV_PI2_4 * dv*dv;
      float alpha = v / fmaxf(1.0f - iou + v, 1e-6f);
      ciou -= alpha * v;
      partial += (1.0f - ciou) * sbls[hit];

      atomicAdd(out + 1, 1.0f);          // num_pos (few hundred atomics total)
    } else if (ioumax <= IGN_THR) {
      partial += bce0(conf);             // no-obj conf BCE (unless ignored)
    }
  }

  // ---- wave32 reduction via V_WMMA_F32_16X16X32_F16 ----
  // A: lane's partial in element 0 -> A[m=lane%16][K=0 or 8]; B: all-ones.
  // D[m][n] = partial[lane m] + partial[lane m+16], replicated over N.
  v16h av; v16h bv;
  #pragma unroll
  for (int k=0;k<16;k++){ av[k] = (_Float16)0.0f; bv[k] = (_Float16)1.0f; }
  av[0] = (_Float16)partial;
  v8f cacc;
  #pragma unroll
  for (int k=0;k<8;k++) cacc[k] = 0.0f;
  cacc = __builtin_amdgcn_wmma_f32_16x16x32_f16(
      /*neg_a=*/false, av, /*neg_b=*/false, bv,
      /*c_mod=*/(short)0, cacc, /*reuse_a=*/false, /*reuse_b=*/false);
  // lanes 0-15 hold rowsums M=0..7, lanes 16-31 hold M=8..15
  float s = cacc[0]+cacc[1]+cacc[2]+cacc[3]+cacc[4]+cacc[5]+cacc[6]+cacc[7];
  s += __shfl_xor(s, 16, 32);            // full wave total on every lane

  const int lane = tid & 31, wave = tid >> 5;
  if (lane == 0) waveSum[wave] = s;
  __syncthreads();
  if (tid == 0) {
    float blockSum = 0.0f;
    #pragma unroll
    for (int w=0; w<8; w++) blockSum += waveSum[w];
    atomicAdd(out, blockSum);
  }
}

__global__ void yolo_finalize(float* __restrict__ out)
{
  out[1] = fmaxf(out[1], 1.0f);          // num_pos = max(sum(obj_mask), 1)
}

extern "C" void kernel_launch(void* const* d_in, const int* in_sizes, int n_in,
                              void* d_out, int out_size, void* d_ws, size_t ws_size,
                              hipStream_t stream)
{
  const float* in  = (const float*)d_in[0];   // (16,255,76,76) fp32
  const float* tgt = (const float*)d_in[1];   // (16,20,5) fp32
  float* out = (float*)d_out;                 // [loss, num_pos]

  hipMemsetAsync(out, 0, 2*sizeof(float), stream);

  dim3 block(256);
  dim3 grid((GRID_WH + 255)/256, 16*A_PER);   // 23 x 48
  yolo_main<<<grid, block, 0, stream>>>(in, tgt, out);
  yolo_finalize<<<1, 1, 0, stream>>>(out);
}
